// VariationalLinearEncoder_46093589021379
// MI455X (gfx1250) — matrix operand order repaired and checked by reading, not compiled, not run on hardware
//
#include <hip/hip_runtime.h>
#include <hip/hip_bf16.h>

typedef __attribute__((ext_vector_type(16))) __bf16 v16bf;
typedef __attribute__((ext_vector_type(8)))  float  v8f;

#define N_NODES 100000
#define IN_CH   256
#define OUT_CH  64
#define HCH     128          // fused mu|logstd output channels
#define NT      8            // N-tiles of 16 cols over HCH
#define KB      (IN_CH / 32) // 8 k-blocks of 32

// ---------------- workspace layout (bytes) ----------------
static constexpr size_t H_ELEMS   = (size_t)N_NODES * HCH;        // 12.8M floats
static constexpr size_t H_BYTES   = H_ELEMS * 4;                  // 51,200,000
static constexpr size_t DINV_OFF  = H_BYTES;                      // N floats
static constexpr size_t DINV_BYTES= (((size_t)N_NODES * 4) + 255) & ~(size_t)255;
static constexpr size_t WPK_OFF   = DINV_OFF + DINV_BYTES;        // 32B aligned
static constexpr size_t WPK_ELEMS = (size_t)NT * KB * 32 * 16;    // 32768 bf16

// ---------------- degree / norm ----------------
__global__ void k_deg_init(float* __restrict__ deg) {
  int n = blockIdx.x * 256 + threadIdx.x;
  if (n < N_NODES) deg[n] = 1.0f;            // self-loop contribution
}

__global__ void k_deg_accum(const long long* __restrict__ dst, int E,
                            float* __restrict__ deg) {
  int e = blockIdx.x * 256 + threadIdx.x;
  if (e < E) atomicAdd(&deg[(int)dst[e]], 1.0f);
}

__global__ void k_dinv(float* __restrict__ deg) {
  int n = blockIdx.x * 256 + threadIdx.x;
  if (n < N_NODES) deg[n] = rsqrtf(deg[n]);  // deg >= 1 always
}

// ---------------- pack W into WMMA B-operand layout (bf16 hi/lo split) -----
// B (32x16, KxN) per 16x16x32 step: lane L holds column N = L&15,
// K = kb*32 + (L>=16 ? 16 : 0) + i, i = 0..15 consecutive.
__global__ void k_packW(const float* __restrict__ W_mu,
                        const float* __restrict__ W_ls,
                        __bf16* __restrict__ wh, __bf16* __restrict__ wl) {
  int t = blockIdx.x * 256 + threadIdx.x;    // one t per (n_tile,kb,lane)
  if (t >= NT * KB * 32) return;
  int lane = t & 31;
  int kb   = (t >> 5) & (KB - 1);
  int nt   = t >> 8;
  int col  = nt * 16 + (lane & 15);
  const float* Wsrc = (col < OUT_CH) ? (W_mu + col) : (W_ls + (col - OUT_CH));
  int kbase = kb * 32 + ((lane >> 4) ? 16 : 0);
#pragma unroll
  for (int i = 0; i < 16; ++i) {
    float  v  = Wsrc[(size_t)(kbase + i) * OUT_CH];   // W[k][col], row-major
    __bf16 hi = (__bf16)v;
    __bf16 lo = (__bf16)(v - (float)hi);
    wh[(size_t)t * 16 + i] = hi;
    wl[(size_t)t * 16 + i] = lo;
  }
}

// ---------------- fused GEMM: h[N x 128] = x @ [W_mu | W_logstd] -----------
// 1 block (8 waves) per 16-row tile; wave w owns N-tile w.
// Split-bf16: x*W ~= xh*Wh + xh*Wl + xl*Wh  (fp32 accumulate).
__global__ __launch_bounds__(256) void k_gemm(
    const float* __restrict__ x,
    const __bf16* __restrict__ wh, const __bf16* __restrict__ wl,
    float* __restrict__ h) {
  __shared__ float xs[16 * 257];             // padded stride -> conflict-free
  const int tid = threadIdx.x;
  const int m0  = blockIdx.x * 16;

  for (int i = tid; i < 16 * IN_CH; i += 256) {
    int r = i >> 8, k = i & (IN_CH - 1);
    xs[r * 257 + k] = x[(size_t)(m0 + r) * IN_CH + k];
  }
  __syncthreads();

  const int wave  = tid >> 5;                // n-tile id
  const int lane  = tid & 31;
  const int row   = lane & 15;               // A-matrix row M
  const int khalf = lane >> 4;               // 0: K 0-15, 1: K 16-31
  const v16bf* Bhi = (const v16bf*)wh;
  const v16bf* Blo = (const v16bf*)wl;

  v8f acc = {};
#pragma unroll
  for (int kb = 0; kb < KB; ++kb) {
    // Build A fragments in contiguous k order (k = 16*khalf + i), same as packed B.
    v16bf a_hi, a_lo;
    const int base = kb * 32 + khalf * 16;
#pragma unroll
    for (int i = 0; i < 8; ++i) {
      float  f  = xs[row * 257 + base + i];
      __bf16 hi = (__bf16)f;
      a_hi[i] = hi;
      a_lo[i] = (__bf16)(f - (float)hi);
    }
#pragma unroll
    for (int i = 0; i < 8; ++i) {
      float  f  = xs[row * 257 + base + 8 + i];
      __bf16 hi = (__bf16)f;
      a_hi[8 + i] = hi;
      a_lo[8 + i] = (__bf16)(f - (float)hi);
    }
    v16bf b_hi = Bhi[(size_t)(wave * KB + kb) * 32 + lane];
    v16bf b_lo = Blo[(size_t)(wave * KB + kb) * 32 + lane];

    acc = __builtin_amdgcn_wmma_f32_16x16x32_bf16(false, a_hi, false, b_hi,
                                                  (short)0, acc, false, false);
    acc = __builtin_amdgcn_wmma_f32_16x16x32_bf16(false, a_hi, false, b_lo,
                                                  (short)0, acc, false, false);
    acc = __builtin_amdgcn_wmma_f32_16x16x32_bf16(false, a_lo, false, b_hi,
                                                  (short)0, acc, false, false);
  }

  // D layout: VGPR r -> M = r (lanes 0-15) / r+8 (lanes 16-31), N = lane&15.
  const int orow0 = m0 + (khalf ? 8 : 0);
  const int col   = wave * 16 + row;
#pragma unroll
  for (int r = 0; r < 8; ++r)
    h[(size_t)(orow0 + r) * HCH + col] = acc[r];
}

// ---------------- bias + self-loop term (fully initializes d_out) ----------
__global__ void k_bias_self(const float* __restrict__ h,
                            const float* __restrict__ dinv,
                            const float* __restrict__ b_mu,
                            const float* __restrict__ b_ls,
                            float* __restrict__ out) {
  int t = blockIdx.x * 256 + threadIdx.x;
  if (t >= N_NODES * OUT_CH) return;
  int n = t >> 6, c = t & 63;
  float w = dinv[n] * dinv[n];               // self-loop norm
  out[t] = b_mu[c] + w * h[(size_t)n * HCH + c];
  out[(size_t)N_NODES * OUT_CH + t] = b_ls[c] + w * h[(size_t)n * HCH + OUT_CH + c];
}

// ---------------- edge gather/scatter: one wave per edge -------------------
__global__ void k_scatter(const long long* __restrict__ src,
                          const long long* __restrict__ dst, int E,
                          const float* __restrict__ h,
                          const float* __restrict__ dinv,
                          float* __restrict__ out) {
  int t = blockIdx.x * 256 + threadIdx.x;
  int e = t >> 5;
  if (e >= E) return;
  int lane = t & 31;
  int s = (int)src[e];
  int d = (int)dst[e];
  float w = dinv[s] * dinv[d];
  // lane covers 4 consecutive channels; 32 lanes span all 128 fused channels
  int c = lane * 4;
  float4 hv = *(const float4*)(h + (size_t)s * HCH + c);
  float* ob = (c < OUT_CH)
                  ? (out + (size_t)d * OUT_CH + c)
                  : (out + (size_t)N_NODES * OUT_CH + (size_t)d * OUT_CH + (c - OUT_CH));
  atomicAdd(ob + 0, w * hv.x);
  atomicAdd(ob + 1, w * hv.y);
  atomicAdd(ob + 2, w * hv.z);
  atomicAdd(ob + 3, w * hv.w);
}

extern "C" void kernel_launch(void* const* d_in, const int* in_sizes, int n_in,
                              void* d_out, int out_size, void* d_ws, size_t ws_size,
                              hipStream_t stream) {
  const float*     x    = (const float*)d_in[0];
  const long long* ei   = (const long long*)d_in[1];   // int64 edge_index
  const float*     W_mu = (const float*)d_in[2];
  const float*     b_mu = (const float*)d_in[3];
  const float*     W_ls = (const float*)d_in[4];
  const float*     b_ls = (const float*)d_in[5];
  float*           out  = (float*)d_out;

  const int E = in_sizes[1] / 2;             // 1,600,000
  const long long* src = ei;
  const long long* dst = ei + E;

  char*   ws   = (char*)d_ws;                // needs ~51.8 MB
  float*  h    = (float*)ws;
  float*  dinv = (float*)(ws + DINV_OFF);    // deg -> dinv in place
  __bf16* wh   = (__bf16*)(ws + WPK_OFF);
  __bf16* wl   = wh + WPK_ELEMS;

  k_deg_init <<<(N_NODES + 255) / 256, 256, 0, stream>>>(dinv);
  k_deg_accum<<<(E + 255) / 256,       256, 0, stream>>>(dst, E, dinv);
  k_dinv     <<<(N_NODES + 255) / 256, 256, 0, stream>>>(dinv);
  k_packW    <<<(NT * KB * 32 + 255) / 256, 256, 0, stream>>>(W_mu, W_ls, wh, wl);
  k_gemm     <<<N_NODES / 16, 256, 0, stream>>>(x, wh, wl, h);
  k_bias_self<<<(N_NODES * OUT_CH + 255) / 256, 256, 0, stream>>>(h, dinv, b_mu, b_ls, out);
  {
    long long threads = (long long)E * 32;
    k_scatter<<<(int)((threads + 255) / 256), 256, 0, stream>>>(src, dst, E, h, dinv, out);
  }
}
